// Attention_cross_28381143892128
// MI455X (gfx1250) — compile-verified, matrix operand order checked by reading, not run on hardware
//
#include <hip/hip_runtime.h>
#include <math.h>

// ---------------------------------------------------------------------------
// CDNA5 (gfx1250) wave32 WMMA types
// ---------------------------------------------------------------------------
typedef __attribute__((ext_vector_type(16))) __bf16 v16bf;
typedef __attribute__((ext_vector_type(8)))  float  v8f;

// K index inside a 16x32 bf16 fragment (ISA 7.12.2): lanes 0-15 hold
// K=0..7 (v0..3) and K=16..23 (v4..7); lanes 16-31 are +8.
__device__ __forceinline__ int kmap(int e, int half) {
  const int vg = e >> 1;
  const int base = (vg < 4) ? (vg << 1) : (16 + ((vg - 4) << 1));
  return base + (e & 1) + (half << 3);
}

// ---------------------------------------------------------------------------
// Generic batched block-GEMM:  C[z] = alpha * A[z] * op(B[z]) (+ bias)
//   block tile 64(M) x 128(N), BK = 64, 256 threads = 8 waves (4Mx2N),
//   wave tile 16x64 -> 4 f32 accumulators, v_wmma_f32_16x16x32_bf16.
//
// LDS tiles are double-buffered and stored pre-swizzled into WMMA fragment
// layout ([frag][lane][16 contiguous bf16]) so each operand is a single
// 32B contiguous LDS read (2x ds_load_b128, conflict-free).  All 10
// fragments of a tile are preloaded before the 8-WMMA burst so DS latency
// overlaps the matrix pipe instead of serializing it.
//
// MODE 0: fp32 store (+bias if non-null)
// MODE 1: bf16 store
// MODE 2: fp32 store + atomic sum/sumsq per batch (instance-norm stats)
// ---------------------------------------------------------------------------
template <typename AT, typename BT, bool TRANSB, int MODE>
__global__ __launch_bounds__(256) void gemm_bf16_wmma(
    const AT* __restrict__ A, long sA, int lda,
    const BT* __restrict__ Bm, long sB, int ldb,
    float* __restrict__ Cf, __bf16* __restrict__ Cb, long sC, int ldc,
    int K, const float* __restrict__ bias, float alpha,
    float* __restrict__ stats)
{
  constexpr int ASZ = 4 * 2 * 32 * 16;   // 4096 bf16 = 8 KB per buffer
  constexpr int BSZ = 2 * 8 * 32 * 16;   // 8192 bf16 = 16 KB per buffer
  __shared__ __align__(32) __bf16 As[2 * ASZ];
  __shared__ __align__(32) __bf16 Bs[2 * BSZ];

  const int z = blockIdx.z;
  A  += (long)z * sA;
  Bm += (long)z * sB;
  const int m0 = blockIdx.y * 64;
  const int n0 = blockIdx.x * 128;
  const int tid  = threadIdx.x;
  const int wave = tid >> 5, lane = tid & 31;
  const int wm = wave >> 1, wn = wave & 1;

  v8f acc[4] = {v8f{}, v8f{}, v8f{}, v8f{}};

  const int nk = K >> 6;
  for (int it = 0; it < nk; ++it) {
    const int kt = it << 6;
    __bf16* Asb = As + (it & 1) * ASZ;
    __bf16* Bsb = Bs + (it & 1) * BSZ;

    // ---- stage A: one fragment-row (16 bf16) per thread, swizzled ----
    {
      const int ls = tid & 31;
      const int frag = tid >> 5;                 // fragM*2 + ks
      const int fm = frag >> 1, ks = frag & 1;
      const long rowb = (long)(m0 + fm * 16 + (ls & 15)) * lda + kt + ks * 32;
      const int hf = ls >> 4;
      v16bf tmp;
#pragma unroll
      for (int t = 0; t < 8; ++t) {              // K pairs are contiguous
        const int k0 = kmap(2 * t, hf);
        tmp[2 * t]     = (__bf16)(float)A[rowb + k0];
        tmp[2 * t + 1] = (__bf16)(float)A[rowb + k0 + 1];
      }
      *(v16bf*)&Asb[tid * 16] = tmp;             // 32B contiguous LDS store
    }
    // ---- stage B: two fragment-rows per thread, swizzled ----
#pragma unroll
    for (int rep = 0; rep < 2; ++rep) {
      const int r = tid + rep * 256;             // 0..511
      const int lr = r & 31;
      const int frag = r >> 5;                   // ks*8 + fragN
      const int ks = frag >> 3, fn = frag & 7;
      const int n = n0 + fn * 16 + (lr & 15);
      const int kb = kt + ks * 32;
      const int hf = lr >> 4;
      v16bf tmp;
#pragma unroll
      for (int e = 0; e < 16; ++e) {
        const int k = kb + kmap(e, hf);
        const float v = TRANSB ? (float)Bm[(long)n * ldb + k]
                               : (float)Bm[(long)k * ldb + n];
        tmp[e] = (__bf16)v;
      }
      *(v16bf*)&Bsb[r * 16] = tmp;
    }
    // L2 prefetch hint for the next K tile (global_prefetch_b8)
    if (kt + 64 < K) {
      __builtin_prefetch((const void*)&A[(long)(m0 + (tid & 63)) * lda + kt + 64], 0, 1);
      __builtin_prefetch(TRANSB
                             ? (const void*)&Bm[(long)(n0 + (tid & 127)) * ldb + kt + 64]
                             : (const void*)&Bm[(long)(kt + 64 + (tid & 63)) * ldb + n0],
                         0, 1);
    }
    __syncthreads();   // single barrier/iter: ping-pong makes read->write safe

    // ---- preload all fragments of this tile, then burst the WMMAs ----
    v16bf af[2];
#pragma unroll
    for (int ks = 0; ks < 2; ++ks)
      af[ks] = *(const v16bf*)&Asb[((wm * 2 + ks) * 32 + lane) * 16];
    v16bf bfr[8];
#pragma unroll
    for (int ks = 0; ks < 2; ++ks)
#pragma unroll
      for (int f = 0; f < 4; ++f)
        bfr[ks * 4 + f] =
            *(const v16bf*)&Bsb[((ks * 8 + wn * 4 + f) * 32 + lane) * 16];
#pragma unroll
    for (int ks = 0; ks < 2; ++ks)
#pragma unroll
      for (int f = 0; f < 4; ++f)
        acc[f] = __builtin_amdgcn_wmma_f32_16x16x32_bf16(
            false, af[ks], false, bfr[ks * 4 + f], (short)0, acc[f],
            false, false);
  }

  // ---- epilogue (C/D layout: col = lane&15, row = r + 8*(lane>>4)) ----
  const int half = lane >> 4, nl = lane & 15;
  float lsum = 0.f, lssq = 0.f;
#pragma unroll
  for (int f = 0; f < 4; ++f) {
    const int col = n0 + wn * 64 + f * 16 + nl;
#pragma unroll
    for (int r = 0; r < 8; ++r) {
      const int row = m0 + wm * 16 + r + half * 8;
      float v = acc[f][r] * alpha;
      if constexpr (MODE == 0) {
        if (bias) v += bias[col];
        Cf[(long)z * sC + (long)row * ldc + col] = v;
      } else if constexpr (MODE == 1) {
        Cb[(long)z * sC + (long)row * ldc + col] = (__bf16)v;
      } else {
        Cf[(long)z * sC + (long)row * ldc + col] = v;
        lsum += v; lssq += v * v;
      }
    }
  }
  if constexpr (MODE == 2) {
#pragma unroll
    for (int o = 16; o > 0; o >>= 1) {
      lsum += __shfl_xor(lsum, o, 32);
      lssq += __shfl_xor(lssq, o, 32);
    }
    if (lane == 0) {
      atomicAdd(&stats[2 * z + 0], lsum);
      atomicAdd(&stats[2 * z + 1], lssq);
    }
  }
}

// ---------------------------------------------------------------------------
// Row L2-normalize fp32 row of length n -> bf16  (q path; F.normalize dim=-1)
// ---------------------------------------------------------------------------
__global__ __launch_bounds__(256) void rownorm_bf16(
    const float* __restrict__ src, __bf16* __restrict__ dst, int n)
{
  const long row = blockIdx.x;
  const float* r = src + row * n;
  float ss = 0.f;
  for (int i = threadIdx.x; i < n; i += 256) { const float v = r[i]; ss += v * v; }
  __shared__ float sh[9];
#pragma unroll
  for (int o = 16; o > 0; o >>= 1) ss += __shfl_xor(ss, o, 32);
  if ((threadIdx.x & 31) == 0) sh[threadIdx.x >> 5] = ss;
  __syncthreads();
  if (threadIdx.x == 0) {
    float tot = 0.f;
    for (int w = 0; w < 8; ++w) tot += sh[w];
    sh[8] = 1.f / fmaxf(sqrtf(tot), 1e-12f);
  }
  __syncthreads();
  const float inv = sh[8];
  for (int i = threadIdx.x; i < n; i += 256)
    dst[row * n + i] = (__bf16)(r[i] * inv);
}

// ---------------------------------------------------------------------------
// 3x3 grouped conv (groups = C/2, 2 in / 2 out per group), SAME zero pad,
// one (out-channel, batch) 32x32 plane per block; optional row L2 norm (k).
// wc points at this branch's [384][2][3][3] weights.
// ---------------------------------------------------------------------------
__global__ __launch_bounds__(256) void gconv3x3_bf16(
    const float* __restrict__ pre,     // [16][384][32][32]
    const float* __restrict__ wc,      // [384][2][3][3]
    __bf16* __restrict__ out,          // [16][384][1024]
    int do_norm)
{
  const int o = blockIdx.x;            // output channel
  const int b = blockIdx.y;            // batch
  const int c0 = (o >> 1) << 1;        // first input channel of the group
  __shared__ float sh[2 * 34 * 34];    // zero-padded halo tiles
  __shared__ float red[9];
  const int tid = threadIdx.x;

  for (int idx = tid; idx < 2 * 34 * 34; idx += 256) {
    const int p = idx / (34 * 34);
    const int rem = idx - p * (34 * 34);
    const int y = rem / 34, x = rem - y * 34;
    const int gy = y - 1, gx = x - 1;
    float v = 0.f;
    if ((unsigned)gy < 32u && (unsigned)gx < 32u)
      v = pre[(((long)b * 384 + c0 + p) * 32 + gy) * 32 + gx];
    sh[idx] = v;
  }
  float w[18];
#pragma unroll
  for (int t = 0; t < 18; ++t) w[t] = wc[o * 18 + t];
  __syncthreads();

  float res[4];
  float ss = 0.f;
#pragma unroll
  for (int q = 0; q < 4; ++q) {
    const int px = tid * 4 + q;
    const int y = px >> 5, x = px & 31;
    float acc = 0.f;
#pragma unroll
    for (int p = 0; p < 2; ++p)
#pragma unroll
      for (int dy = 0; dy < 3; ++dy)
#pragma unroll
        for (int dx = 0; dx < 3; ++dx)
          acc += w[p * 9 + dy * 3 + dx] * sh[p * 34 * 34 + (y + dy) * 34 + (x + dx)];
    res[q] = acc;
    ss += acc * acc;
  }
  float inv = 1.f;
  if (do_norm) {
#pragma unroll
    for (int o2 = 16; o2 > 0; o2 >>= 1) ss += __shfl_xor(ss, o2, 32);
    if ((tid & 31) == 0) red[tid >> 5] = ss;
    __syncthreads();
    if (tid == 0) {
      float tot = 0.f;
      for (int wv = 0; wv < 8; ++wv) tot += red[wv];
      red[8] = 1.f / fmaxf(sqrtf(tot), 1e-12f);
    }
    __syncthreads();
    inv = red[8];
  }
  __bf16* dst = out + ((long)b * 384 + o) * 1024;
#pragma unroll
  for (int q = 0; q < 4; ++q)
    dst[tid * 4 + q] = (__bf16)(res[q] * inv);
}

// ---------------------------------------------------------------------------
// Instance-norm (stats over the whole 384x384 map per batch) + row softmax,
// emit bf16 attn for the next WMMA GEMM. One row per 128-thread block.
// ---------------------------------------------------------------------------
__global__ __launch_bounds__(128) void inorm_softmax_bf16(
    const float* __restrict__ attn,    // [16][384][384]
    const float* __restrict__ stats,   // [16][2]  (sum, sumsq)
    __bf16* __restrict__ outsm)        // [16][384][384]
{
  const int c = blockIdx.x, b = blockIdx.y;
  const float* row = attn + ((long)b * 384 + c) * 384;
  const float inv_n = 1.f / (384.f * 384.f);
  const float mean = stats[2 * b] * inv_n;
  const float var  = stats[2 * b + 1] * inv_n - mean * mean;
  const float istd = rsqrtf(var + 1e-5f);
  const int tid = threadIdx.x;

  float v[3];
  float mx = -3.4e38f;
#pragma unroll
  for (int q = 0; q < 3; ++q) {
    v[q] = (row[q * 128 + tid] - mean) * istd;
    mx = fmaxf(mx, v[q]);
  }
  __shared__ float red[4];
#pragma unroll
  for (int o = 16; o > 0; o >>= 1) mx = fmaxf(mx, __shfl_xor(mx, o, 32));
  if ((tid & 31) == 0) red[tid >> 5] = mx;
  __syncthreads();
  if (tid == 0) {
    float m = red[0];
    for (int w = 1; w < 4; ++w) m = fmaxf(m, red[w]);
    red[0] = m;
  }
  __syncthreads();
  mx = red[0];

  float s = 0.f;
#pragma unroll
  for (int q = 0; q < 3; ++q) { v[q] = __expf(v[q] - mx); s += v[q]; }
#pragma unroll
  for (int o = 16; o > 0; o >>= 1) s += __shfl_xor(s, o, 32);
  __syncthreads();
  if ((tid & 31) == 0) red[tid >> 5] = s;
  __syncthreads();
  if (tid == 0) red[0] = red[0] + red[1] + red[2] + red[3];
  __syncthreads();
  const float invs = 1.f / red[0];

  __bf16* dst = outsm + ((long)b * 384 + c) * 384;
#pragma unroll
  for (int q = 0; q < 3; ++q) dst[q * 128 + tid] = (__bf16)(v[q] * invs);
}

__global__ void zero_f32(float* p, int n) {
  const int i = blockIdx.x * blockDim.x + threadIdx.x;
  if (i < n) p[i] = 0.f;
}

// ---------------------------------------------------------------------------
// Host-side orchestration
// ---------------------------------------------------------------------------
extern "C" void kernel_launch(void* const* d_in, const int* in_sizes, int n_in,
                              void* d_out, int out_size, void* d_ws, size_t ws_size,
                              hipStream_t stream)
{
  (void)in_sizes; (void)n_in; (void)out_size; (void)ws_size;
  const float* emb[4] = {(const float*)d_in[0], (const float*)d_in[1],
                         (const float*)d_in[2], (const float*)d_in[3]};
  const float* text = (const float*)d_in[4];
  const float* wq   = (const float*)d_in[5];
  const float* bq   = (const float*)d_in[6];
  const float* mk   = (const float*)d_in[7];
  const float* mv   = (const float*)d_in[8];
  const float* kw   = (const float*)d_in[9];
  const float* vw   = (const float*)d_in[10];
  const float* po   = (const float*)d_in[11];
  float* out = (float*)d_out;

  const int Bn = 16, C = 384, HW = 1024, TXT = 768;

  char* ws = (char*)d_ws;
  size_t off = 0;
  auto alloc = [&](size_t bytes) {
    void* p = ws + off;
    off = (off + bytes + 255) & ~(size_t)255;
    return p;
  };
  float*  qraw = (float*) alloc((size_t)Bn * C * HW * 4);
  __bf16* qbf  = (__bf16*)alloc((size_t)Bn * C * HW * 2);
  float*  kpre = (float*) alloc((size_t)Bn * C * HW * 4);
  float*  vpre = (float*) alloc((size_t)Bn * C * HW * 4);
  __bf16* kbf  = (__bf16*)alloc((size_t)Bn * C * HW * 2);
  __bf16* vbf  = (__bf16*)alloc((size_t)Bn * C * HW * 2);
  float*  attn = (float*) alloc((size_t)Bn * C * C * 4);
  __bf16* atsm = (__bf16*)alloc((size_t)Bn * C * C * 2);
  __bf16* obf  = (__bf16*)alloc((size_t)Bn * C * HW * 2);
  float*  stat = (float*) alloc((size_t)Bn * 2 * 4);

  const dim3 blk(256);

  // q = l2norm(text @ wq^T + bq): M=B*C=6144, N=HW=1024, K=TXT=768
  gemm_bf16_wmma<float, float, true, 0><<<dim3(8, 96, 1), blk, 0, stream>>>(
      text, 0, TXT, wq, 0, TXT, qraw, nullptr, 0, HW, TXT, bq, 1.f, nullptr);
  rownorm_bf16<<<dim3(Bn * C), blk, 0, stream>>>(qraw, qbf, HW);

  const float scale = 1.f / sqrtf((float)C);
  for (int i = 0; i < 4; ++i) {
    // 1x1 projections: kpre/vpre[b] = mk/mv[i] @ emb[i][b]   (M=C,N=HW,K=C)
    gemm_bf16_wmma<float, float, false, 0><<<dim3(8, 6, 16), blk, 0, stream>>>(
        mk + (size_t)i * C * C, 0, C, emb[i], (long)C * HW, HW,
        kpre, nullptr, (long)C * HW, HW, C, nullptr, 1.f, nullptr);
    gemm_bf16_wmma<float, float, false, 0><<<dim3(8, 6, 16), blk, 0, stream>>>(
        mv + (size_t)i * C * C, 0, C, emb[i], (long)C * HW, HW,
        vpre, nullptr, (long)C * HW, HW, C, nullptr, 1.f, nullptr);

    // grouped 3x3 conv (+ l2norm over HW for k)
    gconv3x3_bf16<<<dim3(C, Bn), blk, 0, stream>>>(kpre, kw + (size_t)i * C * 18, kbf, 1);
    gconv3x3_bf16<<<dim3(C, Bn), blk, 0, stream>>>(vpre, vw + (size_t)i * C * 18, vbf, 0);

    // attn = scale * q @ k^T  (M=C, N=C, K=HW) + instance-norm stats
    zero_f32<<<dim3(1), dim3(64), 0, stream>>>(stat, Bn * 2);
    gemm_bf16_wmma<__bf16, __bf16, true, 2><<<dim3(3, 6, 16), blk, 0, stream>>>(
        qbf, (long)C * HW, HW, kbf, (long)C * HW, HW,
        attn, nullptr, (long)C * C, C, HW, nullptr, scale, stat);

    // softmax(inst_norm(attn)) -> bf16
    inorm_softmax_bf16<<<dim3(C, Bn), dim3(128), 0, stream>>>(attn, stat, atsm);

    // o = attn_sm @ v  (M=C, N=HW, K=C) -> bf16
    gemm_bf16_wmma<__bf16, __bf16, false, 1><<<dim3(8, 6, 16), blk, 0, stream>>>(
        atsm, (long)C * C, C, vbf, (long)C * HW, HW,
        nullptr, obf, (long)C * HW, HW, C, nullptr, 1.f, nullptr);

    // out[i] = po[i] @ o  (M=C, N=HW, K=C) -> fp32 into d_out
    gemm_bf16_wmma<float, __bf16, false, 0><<<dim3(8, 6, 16), blk, 0, stream>>>(
        po + (size_t)i * C * C, 0, C, obf, (long)C * HW, HW,
        out + (size_t)i * Bn * C * HW, nullptr, (long)C * HW, HW, C,
        nullptr, 1.f, nullptr);
  }
}
